// Block_420906795461
// MI455X (gfx1250) — compile-verified, matrix operand order checked by reading, not run on hardware
//
#include <hip/hip_runtime.h>
#include <hip/hip_bf16.h>
#include <math.h>

// ---------------------------------------------------------------------------
// MI455X (gfx1250) fused transformer block.
// Compute-bound (~860 GFLOP vs ~340MB traffic @23.3TB/s => must live on WMMA).
// All GEMMs + attention matmuls use V_WMMA_F32_16X16X32_F16 (f16 in, f32 acc).
// GEMM tiles staged with GLOBAL_LOAD_ASYNC_TO_LDS_B128 (ASYNCcnt-tracked,
// no VGPR round-trip) into double-buffered LDS; K-loop unrolled by two tiles
// so both buffers are statically addressed (no cndmask selection in the hot
// loop). wave32 everywhere; 256-thread blocks = 8 waves.
// ---------------------------------------------------------------------------

typedef _Float16 h16;
typedef __attribute__((ext_vector_type(16))) _Float16     v16h;
typedef __attribute__((ext_vector_type(8)))  float        v8f;
typedef __attribute__((ext_vector_type(4)))  unsigned int u32x4;

#define DIM    2048
#define HEADS  16
#define HD     128
#define SEQ    512
#define NTOK   8192                     // 8 * 2 * 512 tokens
#define QKDIM  (3 * DIM)                // 6144
#define FF     (4 * DIM)                // 8192
#define ATT_SCALE 0.08838834764831845f  // 128^-0.5
#define LN_EPS 1e-5f

// --------------------------- workspace layout ------------------------------
#define XH_OFF    0ull
#define WQKV_OFF  33554432ull
#define WPROJ_OFF 58720256ull
#define W1_OFF    67108864ull
#define W2_OFF    100663296ull
#define QKV_OFF   134217728ull          // reused later as MLP hidden
#define O_OFF     268435456ull
#define X1_OFF    301989888ull

// ---------------------------------------------------------------------------
// WMMA fragment load: 16x32 f16 (A, or B read from an [N][K] surface).
//   lanes 0-15 : row = roff+lane,    K = kb+[0..7] and kb+[16..23]
//   lanes 16-31: row = roff+lane-16, K = kb+[8..15] and kb+[24..31]
// ---------------------------------------------------------------------------
__device__ __forceinline__ v16h frag_ld(const h16* __restrict__ base, int pitch,
                                        int roff, int kb, int lane) {
  const int r  = roff + (lane & 15);
  const int k0 = kb + ((lane >> 4) << 3);
  const h16* p = base + (size_t)r * pitch + k0;
  v16h v;
  ((u32x4*)&v)[0] = *(const u32x4*)p;
  ((u32x4*)&v)[1] = *(const u32x4*)(p + 16);
  return v;
}

__device__ __forceinline__ v8f wmma_f16(v16h a, v16h b, v8f c) {
  return __builtin_amdgcn_wmma_f32_16x16x32_f16(false, a, false, b,
                                                (short)0, c, false, false);
}

// Async copy 16B/lane global -> LDS (VGPR-bypass, ASYNCcnt-tracked).
// Generic LDS pointers on AMDGPU carry the LDS byte address in their low 32b.
__device__ __forceinline__ void async_g2l_b128(unsigned lds_addr,
                                               const h16* g) {
  asm volatile("global_load_async_to_lds_b128 %0, %1, off"
               :: "v"(lds_addr), "v"((unsigned long long)(size_t)g)
               : "memory");
}
#define WAIT_ASYNC_6() asm volatile("s_wait_asynccnt 0x6" ::: "memory")
#define WAIT_ASYNC_0() asm volatile("s_wait_asynccnt 0x0" ::: "memory")

// --------------------------- fp32 -> f16 cast ------------------------------
__global__ __launch_bounds__(256) void cast_f2h(const float* __restrict__ in,
                                                h16* __restrict__ out, int n) {
  int i = blockIdx.x * 256 + threadIdx.x;
  if (i < n) out[i] = (h16)in[i];
}

// --------------------- LayerNorm + cast to f16 (row/block) -----------------
__global__ __launch_bounds__(256) void ln_cast(const float* __restrict__ x,
                                               const float* __restrict__ g,
                                               const float* __restrict__ bta,
                                               h16* __restrict__ out) {
  __shared__ float red[16];
  const int row = blockIdx.x, tid = threadIdx.x;
  const int lane = tid & 31, wid = tid >> 5;
  const float* xr = x + (size_t)row * DIM;
  float v[8], s = 0.f, ss = 0.f;
#pragma unroll
  for (int i = 0; i < 8; ++i) {
    v[i] = xr[tid + i * 256];
    s += v[i]; ss += v[i] * v[i];
  }
#pragma unroll
  for (int off = 16; off > 0; off >>= 1) {
    s  += __shfl_xor(s,  off, 32);
    ss += __shfl_xor(ss, off, 32);
  }
  if (lane == 0) { red[wid] = s; red[8 + wid] = ss; }
  __syncthreads();
  if (wid == 0 && lane < 8) {
    float a = red[lane], b2 = red[8 + lane];
#pragma unroll
    for (int off = 4; off > 0; off >>= 1) {
      a  += __shfl_xor(a,  off, 32);
      b2 += __shfl_xor(b2, off, 32);
    }
    if (lane == 0) { red[0] = a; red[1] = b2; }
  }
  __syncthreads();
  const float mean = red[0] * (1.f / DIM);
  const float var  = red[1] * (1.f / DIM) - mean * mean;
  const float rs   = rsqrtf(var + LN_EPS);
  h16* orow = out + (size_t)row * DIM;
#pragma unroll
  for (int i = 0; i < 8; ++i) {
    int c = tid + i * 256;
    orow[c] = (h16)((v[i] - mean) * rs * g[c] + bta[c]);
  }
}

// ---------------------------------------------------------------------------
// Tiled WMMA GEMM:  C[M,N] = A[M,K] * B[N,K]^T   (both f16 row-major)
// Block tile 64x128, 8 waves of 32x32, K-step 64, double-buffered LDS with
// async global->LDS staging; requires K % 128 == 0. Epilogues:
//   epi 0: f16 store, scale cols < scale_cols by `scale`        (QKV)
//   epi 1: f32 store, + bias[col] + resid[row,col]              (proj / final)
//   epi 2: f16 store, exact GELU(acc + bias[col])               (MLP up)
// ---------------------------------------------------------------------------
__global__ __launch_bounds__(256) void gemm_wmma(
    const h16* __restrict__ A, const h16* __restrict__ B,
    int M, int N, int K,
    h16* __restrict__ outh, float* __restrict__ outf,
    const float* __restrict__ bias, const float* __restrict__ resid,
    int epi, int scale_cols, float scale) {
  __shared__ h16 As[2][64 * 72];
  __shared__ h16 Bs[2][128 * 72];
  const int tid = threadIdx.x, lane = tid & 31, wid = tid >> 5;
  const int wm = (wid >> 2) * 32;          // 0 / 32
  const int wn = (wid & 3) * 32;           // 0..96
  const int bm = blockIdx.y * 64;
  const int bn = blockIdx.x * 128;

  v8f acc[2][2] = {};

  const int ar = tid >> 2, aseg = (tid & 3) * 16;   // A: 64 rows x 64 halves
  const int br = tid >> 1, bseg = (tid & 1) * 32;   // B: 128 rows x 64 halves
  const h16* gpa = A + (size_t)(bm + ar) * K + aseg;
  const h16* gpb = B + (size_t)(bn + br) * K + bseg;
  unsigned ldsA[2], ldsB[2];
  ldsA[0] = (unsigned)(size_t)&As[0][ar * 72 + aseg];
  ldsA[1] = (unsigned)(size_t)&As[1][ar * 72 + aseg];
  ldsB[0] = (unsigned)(size_t)&Bs[0][br * 72 + bseg];
  ldsB[1] = (unsigned)(size_t)&Bs[1][br * 72 + bseg];

  // issue one 64-wide K tile (6 async b128 ops per wave) into buffer `buf`
  auto stage = [&](int kofs, int buf) {
    async_g2l_b128(ldsA[buf],      gpa + kofs);
    async_g2l_b128(ldsA[buf] + 16, gpa + kofs + 8);
#pragma unroll
    for (int q = 0; q < 4; ++q)
      async_g2l_b128(ldsB[buf] + q * 16, gpb + kofs + q * 8);
  };
  // 8 WMMAs over one staged 64-wide K tile
  auto compute = [&](const h16* Asb, const h16* Bsb) {
#pragma unroll
    for (int kk = 0; kk < 64; kk += 32) {
      v16h a0 = frag_ld(Asb, 72, wm,      kk, lane);
      v16h a1 = frag_ld(Asb, 72, wm + 16, kk, lane);
      v16h b0 = frag_ld(Bsb, 72, wn,      kk, lane);
      v16h b1 = frag_ld(Bsb, 72, wn + 16, kk, lane);
      acc[0][0] = wmma_f16(a0, b0, acc[0][0]);
      acc[0][1] = wmma_f16(a0, b1, acc[0][1]);
      acc[1][0] = wmma_f16(a1, b0, acc[1][0]);
      acc[1][1] = wmma_f16(a1, b1, acc[1][1]);
    }
  };

  stage(0, 0);
  for (int kb = 0; kb < K; kb += 128) {
    // ---- tile kb (buf0): overlap with staging tile kb+64 into buf1 ----
    stage(kb + 64, 1);
    __builtin_prefetch(gpa + kb + 128, 0, 3);
    __builtin_prefetch(gpb + kb + 128, 0, 3);
    WAIT_ASYNC_6();          // 12 outstanding -> wait until only newest 6 left
    __syncthreads();
    compute(As[0], Bs[0]);
    __syncthreads();
    // ---- tile kb+64 (buf1): overlap with staging tile kb+128 into buf0 ----
    if (kb + 128 < K) {
      stage(kb + 128, 0);
      WAIT_ASYNC_6();
    } else {
      WAIT_ASYNC_0();
    }
    __syncthreads();
    compute(As[1], Bs[1]);
    __syncthreads();
  }

  const int mlo = (lane >> 4) << 3;   // C/D: M = vgpr + 8*(lane>=16)
  const int nn  = lane & 15;          //      N = lane & 15
#pragma unroll
  for (int mi = 0; mi < 2; ++mi)
#pragma unroll
    for (int ni = 0; ni < 2; ++ni)
#pragma unroll
      for (int r = 0; r < 8; ++r) {
        const size_t row = bm + wm + mi * 16 + mlo + r;
        const size_t col = bn + wn + ni * 16 + nn;
        const float  v   = acc[mi][ni][r];
        if (epi == 0) {
          outh[row * N + col] =
              (h16)(v * (col < (size_t)scale_cols ? scale : 1.f));
        } else if (epi == 1) {
          outf[row * N + col] = v + bias[col] + resid[row * N + col];
        } else {
          float t = v + bias[col];
          outh[row * N + col] =
              (h16)(0.5f * t * (1.f + erff(t * 0.70710678118654752f)));
        }
      }
}

// ---------------------------------------------------------------------------
// Attention per (batch, pair-slot t, head, 16-row q tile):
//   S = Q K^T (WMMA, B-fragments straight from global: K rows contiguous in D)
//   row-wise stable softmax (wave per row, shuffle reductions, v_exp_f32)
//   P rewritten in-place as f16 over its own S row; O = P V with V staged
//   transposed through LDS. Q was pre-scaled by ATT_SCALE in the QKV GEMM.
// ---------------------------------------------------------------------------
__global__ __launch_bounds__(256) void attn_wmma(const h16* __restrict__ qkv,
                                                 h16* __restrict__ o) {
  __shared__ float Sbuf[16 * 512];   // scores; rows reused as f16 P (pitch 1024h)
  __shared__ h16   Qs[16 * 136];
  __shared__ h16   Vt[128 * 72];     // V transposed: [d][j'] for a 64-j chunk

  const int tid = threadIdx.x, lane = tid & 31, wid = tid >> 5;
  const int b = blockIdx.z >> 1, t = blockIdx.z & 1;
  const int h = blockIdx.y;
  const int q0 = blockIdx.x * 16;
  const size_t tokQ0 = (size_t)(b * 2) * SEQ;       // q always from pair slot 0
  const size_t tokK0 = (size_t)(b * 2 + t) * SEQ;
  const int qcol = h * HD;
  const int kcol = DIM + h * HD;
  const int vcol = 2 * DIM + h * HD;

  // stage Q tile 16x128
  {
    const int m = tid >> 4, dseg = (tid & 15) * 8;
    const h16* gp = qkv + (tokQ0 + q0 + m) * QKDIM + qcol + dseg;
    *(u32x4*)&Qs[m * 136 + dseg] = *(const u32x4*)gp;
  }
  __syncthreads();

  // ---- phase 1: S = Q K^T ; wave w owns score cols [w*64, w*64+64) ----
  const h16* Kbase = qkv + tokK0 * QKDIM + kcol;
#pragma unroll
  for (int ni = 0; ni < 4; ++ni) {
    const int n0 = wid * 64 + ni * 16;
    v8f acc = {};
#pragma unroll
    for (int kd = 0; kd < HD; kd += 32) {
      v16h aq = frag_ld(Qs, 136, 0, kd, lane);
      v16h bk = frag_ld(Kbase, QKDIM, n0, kd, lane);
      acc = wmma_f16(aq, bk, acc);
    }
    const int mlo = (lane >> 4) << 3, nn = lane & 15;
#pragma unroll
    for (int r = 0; r < 8; ++r)
      Sbuf[(mlo + r) * 512 + n0 + nn] = acc[r];
  }
  __syncthreads();

  // ---- phase 2: softmax; wave handles rows {2w, 2w+1}; each row is fully
  // in registers before its in-place f16 rewrite -> no cross-wave hazard ----
#pragma unroll
  for (int rr = 0; rr < 2; ++rr) {
    const int row = wid * 2 + rr;
    float vr[16], mx = -1e30f;
#pragma unroll
    for (int i = 0; i < 16; ++i) {
      vr[i] = Sbuf[row * 512 + lane + i * 32];
      mx = fmaxf(mx, vr[i]);
    }
#pragma unroll
    for (int off = 16; off > 0; off >>= 1) mx = fmaxf(mx, __shfl_xor(mx, off, 32));
    float sum = 0.f;
#pragma unroll
    for (int i = 0; i < 16; ++i) { vr[i] = __expf(vr[i] - mx); sum += vr[i]; }
#pragma unroll
    for (int off = 16; off > 0; off >>= 1) sum += __shfl_xor(sum, off, 32);
    const float inv = 1.f / sum;
    h16* prow = (h16*)&Sbuf[row * 512];
#pragma unroll
    for (int i = 0; i < 16; ++i) prow[lane + i * 32] = (h16)(vr[i] * inv);
  }
  __syncthreads();

  // ---- phase 3: O = P V ; wave w owns output cols d in [w*16, w*16+16) ----
  const h16* Vbase = qkv + tokK0 * QKDIM + vcol;
  const h16* Pb = (const h16*)Sbuf;   // pitch 1024 halves, 512 valid cols
  v8f oacc = {};
  for (int jt = 0; jt < SEQ; jt += 64) {
    {   // stage V chunk transposed: V[jt+j'][d] -> Vt[d][j']
      const int jr = tid >> 2, dseg = (tid & 3) * 32;
      const h16* gp = Vbase + (size_t)(jt + jr) * QKDIM + dseg;
      h16 tmp[32];
#pragma unroll
      for (int q = 0; q < 4; ++q)
        *(u32x4*)&tmp[q * 8] = *(const u32x4*)(gp + q * 8);
#pragma unroll
      for (int i = 0; i < 32; ++i) Vt[(dseg + i) * 72 + jr] = tmp[i];
    }
    __syncthreads();
#pragma unroll
    for (int kk = 0; kk < 64; kk += 32) {
      v16h ap = frag_ld(Pb, 1024, 0, jt + kk, lane);
      v16h bv = frag_ld(Vt, 72, wid * 16, kk, lane);
      oacc = wmma_f16(ap, bv, oacc);
    }
    __syncthreads();
  }
  const int mlo = (lane >> 4) << 3, nn = lane & 15;
  h16* obase = o + ((size_t)(b * 2 + t) * SEQ + q0) * DIM + h * HD + wid * 16;
#pragma unroll
  for (int r = 0; r < 8; ++r)
    obase[(size_t)(mlo + r) * DIM + nn] = (h16)oacc[r];
}

// ---------------------------------------------------------------------------
extern "C" void kernel_launch(void* const* d_in, const int* in_sizes, int n_in,
                              void* d_out, int out_size, void* d_ws,
                              size_t ws_size, hipStream_t stream) {
  const float* x      = (const float*)d_in[0];
  const float* gamma1 = (const float*)d_in[1];
  const float* beta1  = (const float*)d_in[2];
  const float* Wqkv   = (const float*)d_in[3];
  const float* Wproj  = (const float*)d_in[4];
  const float* bproj  = (const float*)d_in[5];
  const float* gamma2 = (const float*)d_in[6];
  const float* beta2  = (const float*)d_in[7];
  const float* W1     = (const float*)d_in[8];
  const float* b1     = (const float*)d_in[9];
  const float* W2     = (const float*)d_in[10];
  const float* b2     = (const float*)d_in[11];
  float* out = (float*)d_out;

  char* ws = (char*)d_ws;
  h16*  xh     = (h16*)(ws + XH_OFF);
  h16*  WqkvH  = (h16*)(ws + WQKV_OFF);
  h16*  WprojH = (h16*)(ws + WPROJ_OFF);
  h16*  W1H    = (h16*)(ws + W1_OFF);
  h16*  W2H    = (h16*)(ws + W2_OFF);
  h16*  qkvH   = (h16*)(ws + QKV_OFF);
  h16*  hact   = (h16*)(ws + QKV_OFF);   // reuse: qkv dead after attention
  h16*  oH     = (h16*)(ws + O_OFF);
  float* x1    = (float*)(ws + X1_OFF);

  // --- weight casts fp32 -> f16 ---
  {
    int n;
    n = QKDIM * DIM; cast_f2h<<<(n + 255) / 256, 256, 0, stream>>>(Wqkv, WqkvH, n);
    n = DIM * DIM;   cast_f2h<<<(n + 255) / 256, 256, 0, stream>>>(Wproj, WprojH, n);
    n = FF * DIM;    cast_f2h<<<(n + 255) / 256, 256, 0, stream>>>(W1, W1H, n);
    n = DIM * FF;    cast_f2h<<<(n + 255) / 256, 256, 0, stream>>>(W2, W2H, n);
  }

  // --- LN1 ---
  ln_cast<<<NTOK, 256, 0, stream>>>(x, gamma1, beta1, xh);

  // --- QKV GEMM (q columns pre-scaled by 1/sqrt(D)) ---
  gemm_wmma<<<dim3(QKDIM / 128, NTOK / 64), 256, 0, stream>>>(
      xh, WqkvH, NTOK, QKDIM, DIM, qkvH, nullptr, nullptr, nullptr,
      /*epi=*/0, /*scale_cols=*/DIM, ATT_SCALE);

  // --- attention ---
  attn_wmma<<<dim3(SEQ / 16, HEADS, 16), 256, 0, stream>>>(qkvH, oH);

  // --- proj GEMM + bias + residual(x) -> x1 (fp32) ---
  gemm_wmma<<<dim3(DIM / 128, NTOK / 64), 256, 0, stream>>>(
      oH, WprojH, NTOK, DIM, DIM, nullptr, x1, bproj, x,
      /*epi=*/1, 0, 1.f);

  // --- LN2 ---
  ln_cast<<<NTOK, 256, 0, stream>>>(x1, gamma2, beta2, xh);

  // --- MLP up GEMM + bias + GELU -> hact (f16) ---
  gemm_wmma<<<dim3(FF / 128, NTOK / 64), 256, 0, stream>>>(
      xh, W1H, NTOK, FF, DIM, hact, nullptr, b1, nullptr,
      /*epi=*/2, 0, 1.f);

  // --- MLP down GEMM + bias + residual(x1) -> out (fp32) ---
  gemm_wmma<<<dim3(DIM / 128, NTOK / 64), 256, 0, stream>>>(
      hact, W2H, NTOK, DIM, FF, nullptr, out, b2, x1,
      /*epi=*/1, 0, 1.f);
}
